// Generator_67645734912979
// MI455X (gfx1250) — compile-verified, matrix operand order checked by reading, not run on hardware
//
#include <hip/hip_runtime.h>

typedef __bf16 bf16_t;
typedef __attribute__((ext_vector_type(16))) __bf16        v16bf;
typedef __attribute__((ext_vector_type(8)))  float         v8f;
typedef __attribute__((ext_vector_type(4)))  unsigned int  v4u;

#define LENGTH   12288
#define FILTERS  512
#define BATCH    8
#define TTILE    32                   // 2 M-tiles of 16 rows per wave (B reuse x2)
#define NT_PER_WAVE 8                 // 8 * 16 = 128 filters per wave, 4 waves = 512
#define NCHUNK   (LENGTH / 32)        // 384 k-chunks of 32
#define PACKED_ELEMS ((size_t)FILTERS * LENGTH)          // bf16 elements
#define PACKED_BYTES (PACKED_ELEMS * 2)                  // 12.58 MB

union FragA { v16bf v; unsigned int d[8]; };
union FragB { v16bf v; v4u q[2]; };

// ---------------------------------------------------------------------------
// Pre-pass: K [F][L] f32  ->  packed bf16, WMMA-B-fragment-major:
//   packed[ch][ftile][lane][i] = bf16( K[ftile*16 + (lane&15)]
//                                       [ch*32 + ((i&8)<<1) + ((lane&16)>>1) + (i&7)] )
// ---------------------------------------------------------------------------
__global__ __launch_bounds__(256)
void pack_weights(const float* __restrict__ kern, unsigned short* __restrict__ packed)
{
    const int gid  = blockIdx.x * 256 + threadIdx.x;   // 0 .. 384*32*32-1
    const int lane = gid & 31;
    const int ft   = (gid >> 5) & 31;
    const int ch   = gid >> 10;
    const int f    = ft * 16 + (lane & 15);
    const float* kb = kern + (size_t)f * LENGTH + ch * 32 + ((lane & 16) >> 1);
    unsigned short* dst = packed + (size_t)gid * 16;
    #pragma unroll
    for (int i = 0; i < 16; ++i) {
        const int k = ((i & 8) << 1) + (i & 7);
        unsigned int u = __float_as_uint(kb[k]);
        u += 0x7FFFu + ((u >> 16) & 1u);               // RNE f32 -> bf16
        dst[i] = (unsigned short)(u >> 16);
    }
}

// ---------------------------------------------------------------------------
// Main fused kernel: 32 t-rows x 512 filters per workgroup.
// Per wave per chunk: 12 LDS dwords (2 overlapping A frags) + 8 B frags,
// 16 WMMAs -> 32 FLOP/byte against L2.
// ---------------------------------------------------------------------------
__global__ __launch_bounds__(128)
void conv_dense_fused(const int*   __restrict__ example,
                      const unsigned short* __restrict__ packed,
                      const float* __restrict__ bias,
                      const float* __restrict__ dw,
                      const float* __restrict__ db,
                      float*       __restrict__ out)
{
    // reversed x row, two copies offset by 1 element, +64 zero pad (triangular mask)
    __shared__ unsigned int rev0[(LENGTH + 64) / 2];
    __shared__ unsigned int rev1[(LENGTH + 64) / 2];
    __shared__ float        red[TTILE][4];

    const int tid  = threadIdx.x;
    const int lane = tid & 31;
    const int wave = tid >> 5;
    const int tile = blockIdx.x % (LENGTH / TTILE);
    const int b    = blockIdx.x / (LENGTH / TTILE);
    const int t0   = tile * TTILE;

    // exact bf16 for pixel values {-1, -0.5, 0, 0.5}
    const unsigned short LUT[4] = {0xBF80u, 0xBF00u, 0x0000u, 0x3F00u};
    const int* xrow = example + (size_t)b * LENGTH;
    unsigned short* r0 = (unsigned short*)rev0;
    unsigned short* r1 = (unsigned short*)rev1;
    for (int j = tid; j < LENGTH + 64; j += 128) {
        unsigned short v0 = (j     < LENGTH) ? LUT[xrow[LENGTH - 1 - j] & 3] : (unsigned short)0;
        unsigned short v1 = (j + 1 < LENGTH) ? LUT[xrow[LENGTH - 2 - j] & 3] : (unsigned short)0;
        r0[j] = v0;                 // rev0[j] = xrev[j]
        r1[j] = v1;                 // rev1[j] = xrev[j+1]
    }
    if (tid < TTILE * 4) ((float*)red)[tid] = 0.0f;
    __syncthreads();

    const int m   = lane & 15;
    const int hi8 = (lane & 16) >> 1;          // 0 or 8 (A K-group; C/D M-offset)

    v8f acc[2][NT_PER_WAVE];                   // [M-tile][N-tile], 128 VGPRs
    #pragma unroll
    for (int mt = 0; mt < 2; ++mt)
        #pragma unroll
        for (int nt = 0; nt < NT_PER_WAVE; ++nt) { v8f z = {}; acc[mt][nt] = z; }

    const int nchunks = (t0 + TTILE + 31) >> 5;
    const v4u* packed4 = (const v4u*)packed;
    // rev position of k=hi8, d0=0 for M-tile 0 (rows t0..t0+15); tile 1 is -16
    const int Rbase = (LENGTH - 1) - t0 - m + hi8;

    for (int ch = 0; ch < nchunks; ++ch) {
        const int R   = Rbase + (ch << 5);
        const int sel = R & 1;                       // parity: pick shifted copy
        const int E2  = (R - sel) >> 1;              // aligned dword index (tile 0)
        const unsigned int* U = sel ? rev1 : rev0;

        FragA fa0, fa1;
        #pragma unroll
        for (int v = 0; v < 4; ++v) {
            const unsigned int lo = U[E2 - 8 + v];   // tile1 K=hi8+0..7
            const unsigned int mid = U[E2 + v];      // tile0 K=hi8+0..7 == tile1 K=16+hi8..
            const unsigned int hi  = U[E2 + 8 + v];  // tile0 K=16+hi8+0..7
            fa1.d[v]     = lo;
            fa1.d[4 + v] = mid;
            fa0.d[v]     = mid;
            fa0.d[4 + v] = hi;
        }

        // ---- 8 filter tiles: each B fragment feeds 2 WMMAs (M-tile reuse) ----
        const v4u* pb = packed4 + (((size_t)ch * 32 + wave * 8) * 32 + lane) * 2;
        #pragma unroll
        for (int nt = 0; nt < NT_PER_WAVE; ++nt) {
            FragB fb;
            fb.q[0] = pb[0];
            fb.q[1] = pb[1];
            __builtin_prefetch(pb + 2048, 0, 1);     // next chunk, same tile
            acc[0][nt] = __builtin_amdgcn_wmma_f32_16x16x32_bf16(
                false, fa0.v, false, fb.v, (short)0, acc[0][nt], false, false);
            acc[1][nt] = __builtin_amdgcn_wmma_f32_16x16x32_bf16(
                false, fa1.v, false, fb.v, (short)0, acc[1][nt], false, false);
            pb += 64;                                // next filter tile
        }
    }

    // ---- fused epilogue: relu(y + bias) @ dense_w, reduce over 512 filters ----
    #pragma unroll
    for (int mt = 0; mt < 2; ++mt) {
        float s[8][4];
        #pragma unroll
        for (int r = 0; r < 8; ++r)
            #pragma unroll
            for (int c = 0; c < 4; ++c) s[r][c] = 0.0f;

        #pragma unroll
        for (int nt = 0; nt < NT_PER_WAVE; ++nt) {
            const int f  = wave * 128 + nt * 16 + m;
            const float bv = bias[f];
            const float w0 = dw[f * 4 + 0], w1 = dw[f * 4 + 1];
            const float w2 = dw[f * 4 + 2], w3 = dw[f * 4 + 3];
            #pragma unroll
            for (int r = 0; r < 8; ++r) {
                float z = acc[mt][nt][r] + bv;
                z = z > 0.0f ? z : 0.0f;
                s[r][0] += z * w0;  s[r][1] += z * w1;
                s[r][2] += z * w2;  s[r][3] += z * w3;
            }
        }
        #pragma unroll
        for (int r = 0; r < 8; ++r) {
            const int trow = mt * 16 + r + hi8;      // C/D: M = r + (lane>=16?8:0)
            #pragma unroll
            for (int c = 0; c < 4; ++c) atomicAdd(&red[trow][c], s[r][c]);
        }
    }
    __syncthreads();

    // 128 outputs = 128 threads: one (t, c) each
    {
        const int tl = tid >> 2, c = tid & 3;
        out[((size_t)b * LENGTH + t0 + tl) * 4 + c] = red[tl][c] + db[c];
    }
}

// ---------------------------------------------------------------------------
// Fallback (no workspace): direct f32 weight loads + in-loop bf16 conversion
// ---------------------------------------------------------------------------
__global__ __launch_bounds__(128)
void conv_dense_fused_f32(const int*   __restrict__ example,
                          const float* __restrict__ kern,
                          const float* __restrict__ bias,
                          const float* __restrict__ dw,
                          const float* __restrict__ db,
                          float*       __restrict__ out)
{
    __shared__ bf16_t xlds[LENGTH];
    __shared__ float  red[16][4];

    const int tid  = threadIdx.x;
    const int lane = tid & 31;
    const int wave = tid >> 5;
    const int tile = blockIdx.x % (LENGTH / 16);
    const int b    = blockIdx.x / (LENGTH / 16);
    const int t0   = tile * 16;

    const int* xrow = example + (size_t)b * LENGTH;
    for (int i = tid; i < LENGTH; i += 128)
        xlds[i] = (bf16_t)((float)xrow[i] * 0.5f - 1.0f);
    if (tid < 64) ((float*)red)[tid] = 0.0f;
    __syncthreads();

    const int m   = lane & 15;
    const int hi8 = (lane & 16) >> 1;

    v8f acc[NT_PER_WAVE];
    #pragma unroll
    for (int nt = 0; nt < NT_PER_WAVE; ++nt) { v8f z = {}; acc[nt] = z; }

    const int nchunks = (t0 + 16 + 31) >> 5;
    for (int ch = 0; ch < nchunks; ++ch) {
        const int d0 = ch << 5;
        v16bf a;
        #pragma unroll
        for (int i = 0; i < 16; ++i) {
            const int k   = ((i & 8) << 1) + hi8 + (i & 7);
            const int idx = t0 + m - d0 - k;
            bf16_t av = (bf16_t)0.0f;
            if (idx >= 0) av = xlds[idx];
            a[i] = av;
        }
        #pragma unroll
        for (int nt = 0; nt < NT_PER_WAVE; ++nt) {
            const int f = wave * 128 + nt * 16 + m;
            const float* kb = kern + (size_t)f * LENGTH + d0 + hi8;
            v16bf bf;
            #pragma unroll
            for (int i = 0; i < 16; ++i) {
                const int k = ((i & 8) << 1) + (i & 7);
                bf[i] = (bf16_t)kb[k];
            }
            acc[nt] = __builtin_amdgcn_wmma_f32_16x16x32_bf16(
                false, a, false, bf, (short)0, acc[nt], false, false);
        }
    }

    float s[8][4];
    #pragma unroll
    for (int r = 0; r < 8; ++r)
        #pragma unroll
        for (int c = 0; c < 4; ++c) s[r][c] = 0.0f;
    #pragma unroll
    for (int nt = 0; nt < NT_PER_WAVE; ++nt) {
        const int f  = wave * 128 + nt * 16 + m;
        const float bv = bias[f];
        const float w0 = dw[f * 4 + 0], w1 = dw[f * 4 + 1];
        const float w2 = dw[f * 4 + 2], w3 = dw[f * 4 + 3];
        #pragma unroll
        for (int r = 0; r < 8; ++r) {
            float z = acc[nt][r] + bv;
            z = z > 0.0f ? z : 0.0f;
            s[r][0] += z * w0;  s[r][1] += z * w1;
            s[r][2] += z * w2;  s[r][3] += z * w3;
        }
    }
    #pragma unroll
    for (int r = 0; r < 8; ++r) {
        const int trow = r + hi8;
        #pragma unroll
        for (int c = 0; c < 4; ++c) atomicAdd(&red[trow][c], s[r][c]);
    }
    __syncthreads();
    if (tid < 64) {
        const int tl = tid >> 2, c = tid & 3;
        out[((size_t)b * LENGTH + t0 + tl) * 4 + c] = red[tl][c] + db[c];
    }
}

extern "C" void kernel_launch(void* const* d_in, const int* in_sizes, int n_in,
                              void* d_out, int out_size, void* d_ws, size_t ws_size,
                              hipStream_t stream) {
    const int*   example = (const int*)  d_in[0];
    const float* kern    = (const float*)d_in[1];
    const float* bias    = (const float*)d_in[2];
    const float* dw      = (const float*)d_in[3];
    const float* db      = (const float*)d_in[4];
    float*       out     = (float*)d_out;

    if (d_ws != nullptr && ws_size >= PACKED_BYTES) {
        unsigned short* packed = (unsigned short*)d_ws;
        pack_weights<<<dim3(NCHUNK * 32 * 32 / 256), 256, 0, stream>>>(kern, packed);
        conv_dense_fused<<<dim3(BATCH * (LENGTH / TTILE)), 128, 0, stream>>>(
            example, packed, bias, dw, db, out);
    } else {
        conv_dense_fused_f32<<<dim3(BATCH * (LENGTH / 16)), 128, 0, stream>>>(
            example, kern, bias, dw, db, out);
    }
}
